// ST_Transformer_adaptive_super_gai_new_515396075929
// MI455X (gfx1250) — matrix operand order repaired and checked by reading, not compiled, not run on hardware
//
#include <hip/hip_runtime.h>
#include <hip/hip_bf16.h>
#include <math.h>

// ---------------- types for WMMA ----------------
typedef __attribute__((ext_vector_type(16))) __bf16         v16bf;
typedef __attribute__((ext_vector_type(16))) unsigned short v16u;
typedef __attribute__((ext_vector_type(8)))  float          v8f;

struct __align__(16) U4 { unsigned int x, y, z, w; };
struct __align__(8)  U2 { unsigned int x, y; };
union Frag16 { U4 q[2]; v16u u; v16bf b; };

__device__ __forceinline__ unsigned short f2bf(float f) {
  unsigned int u = __float_as_uint(f);
  u += 0x7FFFu + ((u >> 16) & 1u);   // round-to-nearest-even
  return (unsigned short)(u >> 16);
}
__device__ __forceinline__ unsigned int pack2bf(float lo, float hi) {
  return (unsigned int)f2bf(lo) | ((unsigned int)f2bf(hi) << 16);
}

// order-preserving float<->uint key for atomic max on floats
__device__ __forceinline__ unsigned int fkey(float f) {
  unsigned int u = __float_as_uint(f);
  return (u & 0x80000000u) ? ~u : (u | 0x80000000u);
}
__device__ __forceinline__ float funkey(unsigned int u) {
  unsigned int v = (u & 0x80000000u) ? (u ^ 0x80000000u) : ~u;
  return __uint_as_float(v);
}

#define GT_M 64
#define GT_N 64
#define GT_K 32
#define CDIV(a, b) (((a) + (b) - 1) / (b))

// ---------------- GEMM: Y[M,N] = X[M,K] @ W[K,N] + bias, bf16 WMMA, f32 accum ----------------
// Block: 256 threads = 8 waves; 4 (row) x 2 (col) waves; each wave -> 16x32 output (2 WMMA tiles).
__global__ __launch_bounds__(256) void gemm_bias_wmma(
    const float* __restrict__ X, const float* __restrict__ W,
    const float* __restrict__ bias, float* __restrict__ Y,
    int M, int K, int N)
{
  __shared__ __align__(16) unsigned short As[GT_M][GT_K];  // X tile, row-major
  __shared__ __align__(16) unsigned short Bt[GT_N][GT_K];  // W tile, TRANSPOSED: Bt[col][k]

  const int tid  = threadIdx.x;
  const int wave = tid >> 5;
  const int lane = tid & 31;
  const int wm   = (wave & 3) << 4;   // 0,16,32,48
  const int wnB  = (wave >> 2) << 5;  // 0,32
  const int bm   = blockIdx.y * GT_M;
  const int bn   = blockIdx.x * GT_N;
  const int half = lane >> 4;         // 0 or 1
  const int lr   = lane & 15;
  const bool mOK = (bm + GT_M <= M);
  const bool nOK = (bn + GT_N <= N);

  v8f acc0 = {}, acc1 = {};

  for (int k0 = 0; k0 < K; k0 += GT_K) {
    const bool kOK = (k0 + GT_K <= K);

    // ---- stage A (64x32) ----
    if (mOK && kOK) {
#pragma unroll
      for (int i = 0; i < 2; ++i) {            // 512 float4 / 256 threads
        int item = tid + (i << 8);
        int r = item >> 3, c4 = (item & 7) << 2;
        const float4 f = *(const float4*)(X + (size_t)(bm + r) * K + k0 + c4);
        U2 p; p.x = pack2bf(f.x, f.y); p.y = pack2bf(f.z, f.w);
        *(U2*)&As[r][c4] = p;
        if (k0 + 2 * GT_K <= K)
          __builtin_prefetch(X + (size_t)(bm + r) * K + k0 + GT_K + c4, 0, 3);
      }
    } else {
#pragma unroll
      for (int i = 0; i < 8; ++i) {            // element-wise edge path
        int idx = tid + (i << 8);
        int r = idx >> 5, c = idx & 31;
        int row = bm + r, kk = k0 + c;
        float f = (row < M && kk < K) ? X[(size_t)row * K + kk] : 0.0f;
        As[r][c] = f2bf(f);
      }
    }

    // ---- stage B (32x64 -> transposed 64x32) ----
    if (nOK && kOK) {
#pragma unroll
      for (int i = 0; i < 4; ++i) {            // 1024 k-pairs / 256 threads
        int item = tid + (i << 8);
        int col = item & 63, kp = item >> 6;   // kp in 0..15
        int kk = k0 + (kp << 1);
        float f0 = W[(size_t)kk * N + bn + col];
        float f1 = W[(size_t)(kk + 1) * N + bn + col];
        *(unsigned int*)&Bt[col][kp << 1] = pack2bf(f0, f1);
      }
    } else {
#pragma unroll
      for (int i = 0; i < 8; ++i) {            // 2048 elems edge path
        int idx = tid + (i << 8);
        int col = idx & 63, k = idx >> 6;
        int kk = k0 + k, c = bn + col;
        float f = (kk < K && c < N) ? W[(size_t)kk * N + c] : 0.0f;
        Bt[col][k] = f2bf(f);
      }
    }
    __syncthreads();

    // ---- fragments: contiguous 16B LDS chunks ----
    // A 16x32: lanes 0-15 K 0..15; lanes 16-31 K 16..31 (matches B)
    Frag16 a, b0, b1;
    const U4* ar = (const U4*)&As[wm + lr][0];      // 4 x U4 per row
    a.q[0] = ar[half << 1];
    a.q[1] = ar[(half << 1) + 1];
    // B 32x16: lanes 0-15 K 0..15 of col; lanes 16-31 K 16..31 of col
    const U4* br0 = (const U4*)&Bt[wnB + lr][0];
    const U4* br1 = (const U4*)&Bt[wnB + 16 + lr][0];
    b0.q[0] = br0[half << 1]; b0.q[1] = br0[(half << 1) + 1];
    b1.q[0] = br1[half << 1]; b1.q[1] = br1[(half << 1) + 1];

    acc0 = __builtin_amdgcn_wmma_f32_16x16x32_bf16(false, a.b, false, b0.b, (short)0, acc0, false, false);
    acc1 = __builtin_amdgcn_wmma_f32_16x16x32_bf16(false, a.b, false, b1.b, (short)0, acc1, false, false);
    __syncthreads();
  }

  // C/D layout: vgpr j <-> M = j (lanes 0-15) or 8+j (lanes 16-31); N = lane&15
  const int rbase = bm + wm + (half ? 8 : 0);
#pragma unroll
  for (int sub = 0; sub < 2; ++sub) {
    int col = bn + wnB + (sub << 4) + lr;
    if (col < N) {
      float bv = bias ? bias[col] : 0.0f;
#pragma unroll
      for (int j = 0; j < 8; ++j) {
        int row = rbase + j;
        if (row < M) {
          float v = (sub == 0) ? acc0[j] : acc1[j];
          Y[(size_t)row * N + col] = v + bv;
        }
      }
    }
  }
}

// ---------------- BatchNorm column stats: one block per column ----------------
__global__ __launch_bounds__(256) void colstats_kernel(
    const float* __restrict__ H, int M, int N, float* __restrict__ mu, float* __restrict__ rsig)
{
  __shared__ float ssum[256];
  __shared__ float ssq[256];
  int col = blockIdx.x;
  float s = 0.f, q = 0.f;
  for (int r = threadIdx.x; r < M; r += 256) {
    float v = H[(size_t)r * N + col];
    s += v; q += v * v;
  }
  ssum[threadIdx.x] = s; ssq[threadIdx.x] = q;
  __syncthreads();
  for (int off = 128; off > 0; off >>= 1) {
    if (threadIdx.x < off) {
      ssum[threadIdx.x] += ssum[threadIdx.x + off];
      ssq[threadIdx.x]  += ssq[threadIdx.x + off];
    }
    __syncthreads();
  }
  if (threadIdx.x == 0) {
    float m   = ssum[0] / (float)M;
    float var = ssq[0] / (float)M - m * m;
    mu[col]   = m;
    rsig[col] = rsqrtf(var + 1e-4f);
  }
}

__global__ void bn_elu_kernel(float* __restrict__ H, unsigned long long total, int Ncols,
                              const float* __restrict__ mu, const float* __restrict__ rsig,
                              const float* __restrict__ g, const float* __restrict__ be)
{
  unsigned long long i = (unsigned long long)blockIdx.x * 256ull + threadIdx.x;
  if (i >= total) return;
  int c = (int)(i % (unsigned long long)Ncols);
  float v = (H[i] - mu[c]) * rsig[c] * g[c] + be[c];
  H[i] = v > 0.f ? v : (expf(v) - 1.f);
}

// ---------------- fills ----------------
__global__ void fill_f32_kernel(float* p, float v, unsigned long long n) {
  unsigned long long i = (unsigned long long)blockIdx.x * 256ull + threadIdx.x;
  if (i < n) p[i] = v;
}
__global__ void fill_u32_kernel(unsigned int* p, unsigned int v, unsigned long long n) {
  unsigned long long i = (unsigned long long)blockIdx.x * 256ull + threadIdx.x;
  if (i < n) p[i] = v;
}

// ---------------- edge kernels (TransformerConv) ----------------
__global__ void edge_logits_kernel(const float* __restrict__ q, const float* __restrict__ k,
                                   const int* __restrict__ src, const int* __restrict__ dst,
                                   int E, int D, float scale,
                                   float* __restrict__ logits, unsigned int* __restrict__ segmax)
{
  int e = blockIdx.x * 256 + threadIdx.x;
  if (e >= E) return;
  int sd = dst[e], ss = src[e];
  const float4* qa = (const float4*)(q + (size_t)sd * D);
  const float4* ka = (const float4*)(k + (size_t)ss * D);
  float acc = 0.f;
  int n4 = D >> 2;
  for (int i = 0; i < n4; ++i) {
    float4 a = qa[i], b = ka[i];
    acc = fmaf(a.x, b.x, acc);
    acc = fmaf(a.y, b.y, acc);
    acc = fmaf(a.z, b.z, acc);
    acc = fmaf(a.w, b.w, acc);
  }
  acc *= scale;
  logits[e] = acc;
  atomicMax(&segmax[sd], fkey(acc));
}

__global__ void edge_exp_kernel(const float* __restrict__ logits, const unsigned int* __restrict__ segmax,
                                const int* __restrict__ dst, int E,
                                float* __restrict__ expv, float* __restrict__ segsum)
{
  int e = blockIdx.x * 256 + threadIdx.x;
  if (e >= E) return;
  int sd = dst[e];
  float ev = expf(logits[e] - funkey(segmax[sd]));
  expv[e] = ev;
  atomicAdd(&segsum[sd], ev);
}

__global__ void edge_alpha_kernel(float* __restrict__ alpha, const float* __restrict__ segsum,
                                  const int* __restrict__ dst, int E)
{
  int e = blockIdx.x * 256 + threadIdx.x;
  if (e >= E) return;
  alpha[e] = alpha[e] / segsum[dst[e]];
}

// out[outIdx[e]*D + d] += alpha[e] * data[gatherIdx[e]*D + d]
__global__ void edge_scatter_kernel(const float* __restrict__ alpha, const float* __restrict__ data,
                                    const int* __restrict__ gatherIdx, const int* __restrict__ outIdx,
                                    int E, int D, float* __restrict__ out)
{
  unsigned long long i = (unsigned long long)blockIdx.x * 256ull + threadIdx.x;
  if (i >= (unsigned long long)E * (unsigned long long)D) return;
  int e = (int)(i / (unsigned long long)D);
  int d = (int)(i % (unsigned long long)D);
  float v = alpha[e] * data[(size_t)gatherIdx[e] * D + d];
  atomicAdd(&out[(size_t)outIdx[e] * D + d], v);
}

// ---------------- elementwise combinators ----------------
__global__ void add2_kernel(const float* a, const float* b, float* out, unsigned long long n) {
  unsigned long long i = (unsigned long long)blockIdx.x * 256ull + threadIdx.x;
  if (i < n) out[i] = a[i] + b[i];
}

__global__ void combine_kernel(const float* a, const float* b, float w1, float w2,
                               float* out, unsigned long long n, int relu) {
  unsigned long long i = (unsigned long long)blockIdx.x * 256ull + threadIdx.x;
  if (i >= n) return;
  float v = w1 * a[i] + w2 * b[i];
  if (relu) v = v > 0.f ? v : 0.f;
  out[i] = v;
}

__global__ void combine_strided_kernel(const float* a, const float* b, float w1, float w2,
                                       float* out, int N, int D, int stride, int off) {
  unsigned long long i = (unsigned long long)blockIdx.x * 256ull + threadIdx.x;
  if (i >= (unsigned long long)N * (unsigned long long)D) return;
  int n = (int)(i / (unsigned long long)D);
  int d = (int)(i % (unsigned long long)D);
  out[(size_t)n * stride + off + d] = w1 * a[i] + w2 * b[i];
}

__global__ void copy_strided_kernel(const float* a, float* out, int N, int D, int stride, int off) {
  unsigned long long i = (unsigned long long)blockIdx.x * 256ull + threadIdx.x;
  if (i >= (unsigned long long)N * (unsigned long long)D) return;
  int n = (int)(i / (unsigned long long)D);
  int d = (int)(i % (unsigned long long)D);
  out[(size_t)n * stride + off + d] = a[i];
}

// ---------------- super-resolution ----------------
__global__ void superres_kernel(const float* __restrict__ w, const int* __restrict__ idx,
                                const float* __restrict__ z, float* __restrict__ zs,
                                unsigned long long total, int Dz, int K)
{
  unsigned long long i = (unsigned long long)blockIdx.x * 256ull + threadIdx.x;
  if (i >= total) return;
  unsigned long long row = i / (unsigned long long)Dz;
  int d = (int)(i % (unsigned long long)Dz);
  float acc = 0.f;
  for (int k = 0; k < K; ++k)
    acc = fmaf(w[row * K + k], z[(size_t)idx[row * K + k] * Dz + d], acc);
  zs[i] = acc;
}

__global__ void zmean_kernel(const float* __restrict__ zs, float* __restrict__ z,
                             int N, int Dz, int SUP)
{
  unsigned long long i = (unsigned long long)blockIdx.x * 256ull + threadIdx.x;
  if (i >= (unsigned long long)N * (unsigned long long)Dz) return;
  int n = (int)(i / (unsigned long long)Dz);
  int d = (int)(i % (unsigned long long)Dz);
  float acc = 0.f;
  for (int s = 0; s < SUP; ++s) acc += zs[((size_t)s * N + n) * Dz + d];
  z[i] = acc / (float)SUP;
}

// ---------------- student-t cluster assignment ----------------
__global__ __launch_bounds__(32) void cluster_q_kernel(
    const float* __restrict__ z, const float* __restrict__ cl,
    float* __restrict__ qout, int N, int C, int D)
{
  __shared__ float qs[32];
  __shared__ float qsum;
  int n = blockIdx.x;
  int c = threadIdx.x;
  float val = 0.f;
  if (c < C) {
    float d2 = 0.f;
    const float* zr = z + (size_t)n * D;
    const float* cr = cl + (size_t)c * D;
    for (int i = 0; i < D; ++i) { float t = zr[i] - cr[i]; d2 = fmaf(t, t, d2); }
    val = 1.0f / (1.0f + d2);
  }
  qs[c] = val;
  __syncthreads();
  if (c == 0) {
    float s = 0.f;
    for (int i = 0; i < C; ++i) s += qs[i];
    qsum = s;
  }
  __syncthreads();
  if (c < C) qout[(size_t)n * C + c] = val / qsum;
}

// =================================================================
extern "C" void kernel_launch(void* const* d_in, const int* in_sizes, int n_in,
                              void* d_out, int out_size, void* d_ws, size_t ws_size,
                              hipStream_t stream)
{
  (void)in_sizes; (void)n_in; (void)out_size; (void)ws_size;
  enum { Nn = 20000, Ee = 320000, INF = 3000, F1 = 512, F2 = 128,
         G1 = 128, G2 = 64, Cc = 20, Kk = 6, SUPn = 6, DZ = F2 + G2 };

  const float* x    = (const float*)d_in[0];
  const int*   ei   = (const int*)d_in[1];
  const int*   eip  = (const int*)d_in[2];
  const float* ew   = (const float*)d_in[3];
  const int*   eidx = (const int*)d_in[4];
  const float* enc1W = (const float*)d_in[5];  const float* enc1b  = (const float*)d_in[6];
  const float* enc1g = (const float*)d_in[7];  const float* enc1be = (const float*)d_in[8];
  const float* enc2W = (const float*)d_in[9];  const float* enc2b  = (const float*)d_in[10];
  const float* enc2g = (const float*)d_in[11]; const float* enc2be = (const float*)d_in[12];
  const float* gc1W[4] = {(const float*)d_in[13], (const float*)d_in[15], (const float*)d_in[17], (const float*)d_in[19]};
  const float* gc1B[4] = {(const float*)d_in[14], (const float*)d_in[16], (const float*)d_in[18], (const float*)d_in[20]};
  const float* ch0W[4] = {(const float*)d_in[21], (const float*)d_in[23], (const float*)d_in[25], (const float*)d_in[27]};
  const float* ch0B[4] = {(const float*)d_in[22], (const float*)d_in[24], (const float*)d_in[26], (const float*)d_in[28]};
  const float* gc2W[4] = {(const float*)d_in[29], (const float*)d_in[31], (const float*)d_in[33], (const float*)d_in[35]};
  const float* gc2B[4] = {(const float*)d_in[30], (const float*)d_in[32], (const float*)d_in[34], (const float*)d_in[36]};
  const float* decW = (const float*)d_in[37]; const float* decb  = (const float*)d_in[38];
  const float* decg = (const float*)d_in[39]; const float* decbe = (const float*)d_in[40];
  const float* clus = (const float*)d_in[41];

  const int* s1 = ei;        const int* d1 = ei + Ee;
  const int* s2 = eip;       const int* d2 = eip + Ee;

  // output regions (return order: z, de_feat, q, feat_x, z_super)
  float* outZ  = (float*)d_out;
  float* outDE = outZ  + (size_t)Nn * DZ;
  float* outQ  = outDE + (size_t)Nn * INF;
  float* outFX = outQ  + (size_t)Nn * Cc;
  float* outZS = outFX + (size_t)Nn * F2;

  // workspace layout
  float* wp = (float*)d_ws;
  float* bufA  = wp; wp += (size_t)Nn * 512;   // h1 / fused qkvs
  float* hbuf  = wp; wp += (size_t)Nn * F2;
  float* xh    = wp; wp += (size_t)Nn * F2;
  float* toutA = wp; wp += (size_t)Nn * F2;
  float* toutB = wp; wp += (size_t)Nn * F2;
  float* agg   = wp; wp += (size_t)Nn * F2;
  float* mix1  = wp; wp += (size_t)Nn * F2;
  float* mix2  = wp; wp += (size_t)Nn * F2;
  float* zcat  = wp; wp += (size_t)Nn * DZ;
  float* logits = wp; wp += Ee;
  float* alphaA = wp; wp += Ee;
  float* alphaB = wp; wp += Ee;
  float* segsum = wp; wp += Nn;
  unsigned int* segmax = (unsigned int*)wp; wp += Nn;
  float* muv  = wp; wp += 3072;
  float* rsig = wp; wp += 3072;

  auto gemm = [&](const float* Xp, const float* Wp, const float* bp, float* Yp, int M, int K, int N) {
    dim3 g(CDIV(N, GT_N), CDIV(M, GT_M));
    gemm_bias_wmma<<<g, dim3(256), 0, stream>>>(Xp, Wp, bp, Yp, M, K, N);
  };
  auto fb_bn = [&](float* H, int M, int N, const float* g, const float* be) {
    colstats_kernel<<<N, 256, 0, stream>>>(H, M, N, muv, rsig);
    unsigned long long tot = (unsigned long long)M * N;
    bn_elu_kernel<<<CDIV(tot, 256ull), 256, 0, stream>>>(H, tot, N, muv, rsig, g, be);
  };
  auto fillf = [&](float* p, float v, unsigned long long n) {
    fill_f32_kernel<<<CDIV(n, 256ull), 256, 0, stream>>>(p, v, n);
  };
  auto tconv_pass = [&](const float* qb, const float* kb, const float* vb, const float* sb,
                        const int* src, const int* dst, int D, float* alpha, float* outb) {
    fill_u32_kernel<<<CDIV((unsigned long long)Nn, 256ull), 256, 0, stream>>>(segmax, 0u, Nn);
    fillf(segsum, 0.f, Nn);
    fillf(agg, 0.f, (unsigned long long)Nn * D);
    float scale = 1.0f / sqrtf((float)D);
    edge_logits_kernel<<<CDIV(Ee, 256), 256, 0, stream>>>(qb, kb, src, dst, Ee, D, scale, logits, segmax);
    edge_exp_kernel<<<CDIV(Ee, 256), 256, 0, stream>>>(logits, segmax, dst, Ee, alpha, segsum);
    edge_alpha_kernel<<<CDIV(Ee, 256), 256, 0, stream>>>(alpha, segsum, dst, Ee);
    unsigned long long tot = (unsigned long long)Ee * D;
    edge_scatter_kernel<<<CDIV(tot, 256ull), 256, 0, stream>>>(alpha, vb, src, dst, Ee, D, agg);
    add2_kernel<<<CDIV((unsigned long long)Nn * D, 256ull), 256, 0, stream>>>(agg, sb, outb, (unsigned long long)Nn * D);
  };
  auto dmix = [&](const float* alpha, const int* src, const int* dst, const float* hsrc, int D, float* out) {
    fillf(out, 0.f, (unsigned long long)Nn * D);
    unsigned long long tot = (unsigned long long)Ee * D;
    edge_scatter_kernel<<<CDIV(tot, 256ull), 256, 0, stream>>>(alpha, hsrc, dst, src, Ee, D, out);
  };

  // ---- encoder ----
  gemm(x, enc1W, enc1b, bufA, Nn, INF, F1);
  fb_bn(bufA, Nn, F1, enc1g, enc1be);
  gemm(bufA, enc2W, enc2b, outFX, Nn, F1, F2);      // feat_x lives in d_out
  fb_bn(outFX, Nn, F2, enc2g, enc2be);

  // ---- gc1 (projections shared by both graphs) ----
  float* qb = bufA;
  float* kb = bufA + (size_t)Nn * G1;
  float* vb = bufA + (size_t)2 * Nn * G1;
  float* sb = bufA + (size_t)3 * Nn * G1;
  for (int j = 0; j < 4; ++j)
    gemm(outFX, gc1W[j], gc1B[j], bufA + (size_t)j * Nn * G1, Nn, F2, G1);
  tconv_pass(qb, kb, vb, sb, s1, d1, G1, alphaA, toutA);
  tconv_pass(qb, kb, vb, sb, s2, d2, G1, alphaB, toutB);
  dmix(alphaA, s1, d1, toutA, G1, mix1);
  dmix(alphaB, s2, d2, toutB, G1, mix2);
  combine_kernel<<<CDIV((unsigned long long)Nn * G1, 256ull), 256, 0, stream>>>(
      mix1, mix2, 0.5f, 0.5f, hbuf, (unsigned long long)Nn * G1, 1);

  // ---- ch0: call 1 on hbuf (graph 1), call 2 on xh (graph 2) ----
  for (int j = 0; j < 4; ++j)
    gemm(hbuf, ch0W[j], ch0B[j], bufA + (size_t)j * Nn * G1, Nn, G1, G1);
  tconv_pass(qb, kb, vb, sb, s1, d1, G1, alphaA, xh);
  for (int j = 0; j < 4; ++j)
    gemm(xh, ch0W[j], ch0B[j], bufA + (size_t)j * Nn * G1, Nn, G1, G1);
  tconv_pass(qb, kb, vb, sb, s2, d2, G1, alphaB, toutB);
  dmix(alphaA, s1, d1, xh, G1, mix1);
  dmix(alphaB, s2, d2, toutB, G1, mix2);
  combine_kernel<<<CDIV((unsigned long long)Nn * G1, 256ull), 256, 0, stream>>>(
      mix1, mix2, 0.5f, 0.5f, hbuf, (unsigned long long)Nn * G1, 1);

  // ---- gc2 -> mu (no relu), projections shared ----
  float* qb2 = bufA;
  float* kb2 = bufA + (size_t)Nn * G2;
  float* vb2 = bufA + (size_t)2 * Nn * G2;
  float* sb2 = bufA + (size_t)3 * Nn * G2;
  for (int j = 0; j < 4; ++j)
    gemm(hbuf, gc2W[j], gc2B[j], bufA + (size_t)j * Nn * G2, Nn, G1, G2);
  tconv_pass(qb2, kb2, vb2, sb2, s1, d1, G2, alphaA, toutA);
  tconv_pass(qb2, kb2, vb2, sb2, s2, d2, G2, alphaB, toutB);
  dmix(alphaA, s1, d1, toutA, G2, mix1);
  dmix(alphaB, s2, d2, toutB, G2, mix2);

  // ---- z = concat(feat_x, mu) ----
  combine_strided_kernel<<<CDIV((unsigned long long)Nn * G2, 256ull), 256, 0, stream>>>(
      mix1, mix2, 0.5f, 0.5f, zcat, Nn, G2, DZ, F2);
  copy_strided_kernel<<<CDIV((unsigned long long)Nn * F2, 256ull), 256, 0, stream>>>(
      outFX, zcat, Nn, F2, DZ, 0);

  // ---- super-resolution ----
  unsigned long long zsTot = (unsigned long long)SUPn * Nn * DZ;
  superres_kernel<<<CDIV(zsTot, 256ull), 256, 0, stream>>>(ew, eidx, zcat, outZS, zsTot, DZ, Kk);
  zmean_kernel<<<CDIV((unsigned long long)Nn * DZ, 256ull), 256, 0, stream>>>(outZS, outZ, Nn, DZ, SUPn);

  // ---- decoder ----
  gemm(outZ, decW, decb, outDE, Nn, DZ, INF);
  fb_bn(outDE, Nn, INF, decg, decbe);

  // ---- cluster assignment ----
  cluster_q_kernel<<<Nn, 32, 0, stream>>>(outZ, clus, outQ, Nn, Cc, DZ);
}
